// deform_DFN_42442866819722
// MI455X (gfx1250) — compile-verified, matrix operand order checked by reading, not run on hardware
//
#include <hip/hip_runtime.h>
#include <hip/hip_bf16.h>
#include <stdint.h>

// ---------------------------------------------------------------------------
// CDNA5 (gfx1250) recurrent deformable-conv predictor.
// Conv layers -> fully-templated NHWC implicit-im2col GEMM on
// v_wmma_f32_16x16x32_f16 (wave32). K is ordered tap-major / channel-minor
// (k = tap*CP + ch, CP = pow2 channel pad): a WMMA f16 pair is one aligned
// b32 load; weight quads are global_load_b128. Branch-free clamped gathers.
// Split-K slices accumulate via global_atomic_add_f32 for occupancy.
// Small layers stage their weight slice into LDS with
// global_load_async_to_lds_b128 (ASYNCcnt) and read B via ds_load_b128.
// ---------------------------------------------------------------------------

typedef __attribute__((ext_vector_type(16))) _Float16 v16h;
typedef __attribute__((ext_vector_type(8)))  float    v8f;

#define SLOPE 0.01f
#define BN_EPS 1e-5f
#define NB 16
#define NUM_IN 10
#define NUM_OUT 10

union AF { v16h v; _Float16 h[16]; uint32_t u[8]; };

__device__ __forceinline__ int iminc(int a, int b) { return a < b ? a : b; }
__device__ __forceinline__ int imaxc(int a, int b) { return a > b ? a : b; }

// Branch-free NHWC im2col gather of one f16 pair (channels ci, ci+1 of one
// pixel). X layout: [b][y][x][CP] f16, padded channels pre-zeroed.
template<int CP, int IH, int IW, int KDIM, int STRIDE, int PAD, int TR, int KPAD>
__device__ __forceinline__ uint32_t fetch_pair(const _Float16* __restrict__ X,
                                               int b, int oy, int ox, int k) {
  constexpr int KK  = KDIM * KDIM;
  constexpr int LCP = (CP == 2) ? 1 : (CP == 16) ? 4 : (CP == 32) ? 5 : (CP == 64) ? 6 : 7;
  static_assert(CP == (1 << LCP), "CP must be a power of two");
  int ci = k & (CP - 1);                       // even (k is even)
  int rr = (unsigned)k >> LCP;                 // tap index
  bool ok = true;
  if constexpr (KPAD != KK * CP) { ok = rr < KK; rr = iminc(rr, KK - 1); }
  int ky = rr / KDIM, kx = rr - ky * KDIM;     // constant divisor
  int iy, ix;
  if constexpr (TR == 0) {
    iy = oy * STRIDE - PAD + ky;
    ix = ox * STRIDE - PAD + kx;
    ok = ok & ((unsigned)iy < (unsigned)IH) & ((unsigned)ix < (unsigned)IW);
  } else {
    int ty = oy + PAD - ky, tx = ox + PAD - kx;      // STRIDE == 2 for all TR layers
    ok = ok & (ty >= 0) & (tx >= 0) & ((ty & 1) == 0) & ((tx & 1) == 0);
    iy = ty >> 1; ix = tx >> 1;
    ok = ok & (iy < IH) & (ix < IW);
  }
  int iyC = iminc(imaxc(iy, 0), IH - 1);
  int ixC = iminc(imaxc(ix, 0), IW - 1);
  uint32_t v = *(const uint32_t*)(X + (((size_t)b * IH + iyC) * IW + ixC) * CP + ci);
  return ok ? v : 0u;
}

// One wave per 16 x NPAD C tile (NPAD == 16*NT). grid = (SK, M/16); M%16==0.
// SK>1: K-slices accumulate into pre-zeroed C with global_atomic_add_f32.
// LDSW: stage this wave's weight slice [NPAD][KPAD/SK] into LDS via
// global_load_async_to_lds_b128, then serve B fragments from LDS.
template<int CP, int IH, int IW, int OH, int OW, int KDIM, int STRIDE, int PAD,
         int TR, int KPAD, int NPAD, int NT, int SK, int LDSW>
__global__ __launch_bounds__(32)
void wmma_conv_kernel(const _Float16* __restrict__ X,
                      const _Float16* __restrict__ W,   // [NPAD][KPAD] f16, zero padded
                      float* __restrict__ C) {          // [M][NPAD] f32
  static_assert(NPAD == 16 * NT, "wave spans full NPAD");
  static_assert((KPAD / SK) % 32 == 0, "K slice must be a multiple of 32");
  constexpr int KS = KPAD / SK;
  const int lane = threadIdx.x & 31;
  const int half = lane >> 4;              // K-half selector per ISA A layout
  const int l16  = lane & 15;
  const int mtile = blockIdx.y << 4;
  const int m = mtile + l16;
  const int kbeg = blockIdx.x * KS;

  __shared__ __align__(16) _Float16 Wlds[LDSW ? (NPAD * KS) : 1];
  if constexpr (LDSW) {
    // Stage W[0..NPAD)[kbeg..kbeg+KS) into LDS, 16B per async chunk.
    constexpr int CHUNKS = NPAD * KS / 8;          // 8 f16 per chunk
    constexpr int CPR    = KS / 8;                 // chunks per row
    for (int c = lane; c < CHUNKS; c += 32) {
      int n = c / CPR, o = c - n * CPR;
      uint32_t ldsoff = (uint32_t)(uintptr_t)(&Wlds[n * KS + o * 8]);
      uint64_t gaddr  = (uint64_t)(uintptr_t)(W + (size_t)n * KPAD + kbeg + o * 8);
      asm volatile("global_load_async_to_lds_b128 %0, %1, off"
                   :: "v"(ldsoff), "v"(gaddr) : "memory");
    }
    asm volatile("s_wait_asynccnt 0" ::: "memory");  // fill done before ds reads
  }

  constexpr int SP = OH * OW;
  int b  = m / SP;       int r  = m - b * SP;
  int oy = r / OW;       int ox = r - oy * OW;

  int kofs[8];
#pragma unroll
  for (int j = 0; j < 8; ++j)
    kofs[j] = ((j < 4) ? (2 * j) : (16 + 2 * (j - 4))) + 8 * half;

  v8f acc[NT];
#pragma unroll
  for (int t = 0; t < NT; ++t) acc[t] = (v8f){0.f,0.f,0.f,0.f,0.f,0.f,0.f,0.f};

  const _Float16* Wrow = W + (size_t)l16 * KPAD;   // global path (LDSW == 0)

  for (int k0 = kbeg; k0 < kbeg + KS; k0 += 32) {
    AF a;
#pragma unroll
    for (int j = 0; j < 8; ++j)
      a.u[j] = fetch_pair<CP,IH,IW,KDIM,STRIDE,PAD,TR,KPAD>(X, b, oy, ox, k0 + kofs[j]);
#pragma unroll
    for (int t = 0; t < NT; ++t) {
      AF w;
#pragma unroll
      for (int j = 0; j < 8; ++j) {
        if constexpr (LDSW)
          w.u[j] = *(const uint32_t*)(&Wlds[(size_t)(t * 16 + l16) * KS + (k0 - kbeg) + kofs[j]]);
        else
          w.u[j] = *(const uint32_t*)(Wrow + (size_t)t * 16 * KPAD + k0 + kofs[j]);
      }
      acc[t] = __builtin_amdgcn_wmma_f32_16x16x32_f16(
          false, a.v, false, w.v, (short)0, acc[t], false, false);
    }
  }
#pragma unroll
  for (int t = 0; t < NT; ++t)
#pragma unroll
    for (int rr = 0; rr < 8; ++rr) {
      int row = mtile + rr + 8 * half;     // C layout: vgpr rr -> M = rr + 8*half
      float* dst = &C[(size_t)row * NPAD + t * 16 + l16];
      if constexpr (SK > 1) unsafeAtomicAdd(dst, acc[t][rr]);
      else                  *dst = acc[t][rr];
    }
}

// ---- BatchNorm(train) fused with bias + LeakyReLU -------------------------
__global__ void bn_stats_kernel(const float* __restrict__ C, const float* __restrict__ bias,
                                float* __restrict__ stats, int M, int Npad) {
  int c = blockIdx.x;
  float bi = bias[c];
  float s = 0.f, ss = 0.f;
  for (int m = threadIdx.x; m < M; m += blockDim.x) {
    float v = C[(size_t)m * Npad + c] + bi;
    v = (v >= 0.f) ? v : SLOPE * v;
    s += v; ss += v * v;
  }
  __shared__ float sh0[256], sh1[256];
  int tid = threadIdx.x;
  sh0[tid] = s; sh1[tid] = ss;
  __syncthreads();
  for (int w = 128; w > 0; w >>= 1) {
    if (tid < w) { sh0[tid] += sh0[tid + w]; sh1[tid] += sh1[tid + w]; }
    __syncthreads();
  }
  if (tid == 0) {
    float mean = sh0[0] / (float)M;
    stats[2 * c]     = mean;
    stats[2 * c + 1] = sh1[0] / (float)M - mean * mean;
  }
}

// Yf (optional): NCHW f32.  Yh (optional): NHWC f16 with channel pad CPH
// (padded channels pre-zeroed once per launch).
__global__ void bn_apply_kernel(const float* __restrict__ C, const float* __restrict__ bias,
                                const float* __restrict__ stats,
                                const float* __restrict__ gamma, const float* __restrict__ beta,
                                float* __restrict__ Yf, _Float16* __restrict__ Yh,
                                int M, int Npad, int Cout, int HW, int CPH) {
  int idx = blockIdx.x * blockDim.x + threadIdx.x;
  if (idx >= M * Cout) return;
  int m = idx / Cout, c = idx - m * Cout;
  float v = C[(size_t)m * Npad + c] + bias[c];
  v = (v >= 0.f) ? v : SLOPE * v;
  v = gamma[c] * (v - stats[2 * c]) * rsqrtf(stats[2 * c + 1] + BN_EPS) + beta[c];
  if (Yh) Yh[(size_t)m * CPH + c] = (_Float16)v;          // NHWC (m = b*HW+p)
  if (Yf) {
    int b = m / HW, p = m - b * HW;
    Yf[((size_t)b * Cout + c) * HW + p] = v;              // NCHW
  }
}

// ---- elementwise / tail kernels -------------------------------------------
__global__ void fill_zero_f32(float* p, int n) {
  int i = blockIdx.x * blockDim.x + threadIdx.x;
  if (i < n) p[i] = 0.f;
}
__global__ void fill_zero_f16(_Float16* p, int n) {
  int i = blockIdx.x * blockDim.x + threadIdx.x;
  if (i < n) p[i] = (_Float16)0.f;
}
// weights -> f16 [NPAD][KPAD], K ordered tap*CP + ci
__global__ void repack_fwd_kernel(const float* w, _Float16* Wb, int Cout, int Cin,
                                  int KK, int CP, int Kpad) {
  int idx = blockIdx.x * blockDim.x + threadIdx.x;
  int total = Cout * Cin * KK;
  if (idx >= total) return;
  int n = idx / (Cin * KK); int r2 = idx - n * (Cin * KK);
  int ci = r2 / KK;         int rr = r2 - ci * KK;        // (cout,cin,kh,kw) flat
  Wb[(size_t)n * Kpad + rr * CP + ci] = (_Float16)w[idx];
}
__global__ void repack_tr_kernel(const float* w, _Float16* Wb, int Cin, int Cout,
                                 int KK, int CP, int Kpad) {
  int idx = blockIdx.x * blockDim.x + threadIdx.x;
  int total = Cin * Cout * KK;
  if (idx >= total) return;
  int ci = idx / (Cout * KK); int r2 = idx - ci * (Cout * KK);
  int co = r2 / KK;           int rr = r2 - co * KK;      // (cin,cout,kh,kw) flat
  Wb[(size_t)co * Kpad + rr * CP + ci] = (_Float16)w[idx];
}
__global__ void frame_prep_kernel(const float* __restrict__ inp, const float* __restrict__ pred,
                                  float* __restrict__ frame, int t) {
  int i = blockIdx.x * blockDim.x + threadIdx.x;
  if (i >= NB * 4096) return;
  frame[i] = (t < NUM_IN) ? inp[(size_t)i * NUM_IN + t] : pred[i];  // (B,1,64,64,T)
}
// x0h: NHWC with CP=2 (ch0 = frame+hs, ch1 = 0)
__global__ void add2_h_kernel(const float* __restrict__ a, const float* __restrict__ b,
                              _Float16* __restrict__ c, int n) {
  int i = blockIdx.x * blockDim.x + threadIdx.x;
  if (i < n) {
    c[2 * i]     = (_Float16)(a[i] + b[i]);
    c[2 * i + 1] = (_Float16)0.f;
  }
}
__global__ void add_inplace_kernel(float* a, const float* b, int n) {
  int i = blockIdx.x * blockDim.x + threadIdx.x;
  if (i < n) a[i] += b[i];
}
__global__ void upsample_kernel(const float* __restrict__ src, float* __restrict__ dstf,
                                _Float16* __restrict__ dsth) {
  int idx = blockIdx.x * blockDim.x + threadIdx.x;        // (B*27,64,64)
  if (idx >= NB * 27 * 4096) return;
  int ox = idx & 63, oy = (idx >> 6) & 63, bc = idx >> 12;
  int sy = (oy * 57) >> 6, sx = (ox * 57) >> 6;           // torch 'nearest'
  float v = src[(size_t)bc * 3249 + sy * 57 + sx];
  dstf[idx] = v;                                          // NCHW f32
  int b = bc / 27, c = bc - 27 * b;
  dsth[((size_t)b * 4096 + (oy * 64 + ox)) * 32 + c] = (_Float16)v;  // NHWC-32
}
__global__ void softmax_kernel(const float* __restrict__ h, float* __restrict__ df) {
  int idx = blockIdx.x * blockDim.x + threadIdx.x;        // (b,p)
  if (idx >= NB * 4096) return;
  int b = idx >> 12, p = idx & 4095;
  const float* base = h + (size_t)b * 27 * 4096 + p;
  float mx = -1e30f;
#pragma unroll
  for (int c = 0; c < 27; ++c) mx = fmaxf(mx, base[c * 4096]);
  float e[27], s = 0.f;
#pragma unroll
  for (int c = 0; c < 27; ++c) { e[c] = __expf(base[c * 4096] - mx); s += e[c]; }
  float inv = 1.f / s;
  float* ob = df + (size_t)b * 27 * 4096 + p;
#pragma unroll
  for (int c = 0; c < 27; ++c) ob[c * 4096] = e[c] * inv;
}
__global__ void offset_kernel(const float* __restrict__ frame, const float* __restrict__ df,
                              float* __restrict__ offx, float* __restrict__ offy) {
  int idx = blockIdx.x * blockDim.x + threadIdx.x;        // (b,n<9,p)
  if (idx >= NB * 9 * 4096) return;
  int p = idx & 4095, n = (idx >> 12) % 9, b = idx / (9 * 4096);
  int y = p >> 6, x = p & 63;
  int ky = n / 3, kx = n - 3 * (n / 3);
  int yy = y + ky - 1, xx = x + kx - 1;
  float u = ((unsigned)yy < 64u && (unsigned)xx < 64u) ? frame[b * 4096 + yy * 64 + xx] : 0.f;
  offx[idx] = u * df[((size_t)b * 27 + 9 + n) * 4096 + p];
  offy[idx] = u * df[((size_t)b * 27 + 18 + n) * 4096 + p];
}
__device__ __forceinline__ float samp_pad(const float* frame, int b, int ix, int iy) {
  if (ix < 1 || ix > 64 || iy < 1 || iy > 64) return 0.f; // zero-pad(1) image
  return frame[b * 4096 + (ix - 1) * 64 + (iy - 1)];
}
__global__ void deform_kernel(const float* __restrict__ frame, const float* __restrict__ offx,
                              const float* __restrict__ offy, float* __restrict__ img) {
  int idx = blockIdx.x * blockDim.x + threadIdx.x;        // (b,n,i,j)
  if (idx >= NB * 9 * 4096) return;
  int p = idx & 4095, n = (idx >> 12) % 9, b = idx / (9 * 4096);
  int i = p >> 6, j = p & 63;
  int a = n / 3, bb = n - 3 * a;
  float px = (float)(i + 1) + (float)(a - 1) + offx[idx];
  float py = (float)(j + 1) + (float)(bb - 1) + offy[idx];
  float fx = floorf(px), fy = floorf(py);
  float qx0 = fminf(fmaxf(fx, 0.f), 65.f),       qx1 = fminf(fmaxf(fx + 1.f, 0.f), 65.f);
  float qy0 = fminf(fmaxf(fy, 0.f), 65.f),       qy1 = fminf(fmaxf(fy + 1.f, 0.f), 65.f);
  float pxc = fminf(fmaxf(px, 0.f), 65.f),       pyc = fminf(fmaxf(py, 0.f), 65.f);
  float g_lt = (1.f + (qx0 - pxc)) * (1.f + (qy0 - pyc));
  float g_rb = (1.f - (qx1 - pxc)) * (1.f - (qy1 - pyc));
  float g_lb = (1.f + (qx0 - pxc)) * (1.f - (qy1 - pyc));
  float g_rt = (1.f - (qx1 - pxc)) * (1.f + (qy0 - pyc));
  int ix0 = (int)qx0, ix1 = (int)qx1, iy0 = (int)qy0, iy1 = (int)qy1;
  float val = g_lt * samp_pad(frame, b, ix0, iy0) + g_rb * samp_pad(frame, b, ix1, iy1)
            + g_lb * samp_pad(frame, b, ix0, iy1) + g_rt * samp_pad(frame, b, ix1, iy0);
  img[(size_t)b * 36864 + (3 * i + a) * 192 + (3 * j + bb)] = val;
}
__global__ void predict_kernel(const float* __restrict__ img, const float* __restrict__ df,
                               float* __restrict__ pred, float* __restrict__ out, int t) {
  int idx = blockIdx.x * blockDim.x + threadIdx.x;        // (b,i,j)
  if (idx >= NB * 4096) return;
  int b = idx >> 12, p = idx & 4095;
  int i = p >> 6, j = p & 63;
  float s = 0.f;
#pragma unroll
  for (int ky = 0; ky < 3; ++ky)
#pragma unroll
    for (int kx = 0; kx < 3; ++kx) {
      int rr = 3 * i + ky - 1, cc = 3 * j + kx - 1;
      float v = ((unsigned)rr < 192u && (unsigned)cc < 192u)
                ? img[(size_t)b * 36864 + rr * 192 + cc] : 0.f;
      s += v * df[((size_t)b * 27 + (ky * 3 + kx)) * 4096 + p];
    }
  pred[idx] = s;
  if (t >= NUM_IN - 1)
    out[(size_t)idx * NUM_OUT + (t - (NUM_IN - 1))] = s;  // (B,1,64,64,T) layout
}

// ---------------------------------------------------------------------------
extern "C" void kernel_launch(void* const* d_in, const int* in_sizes, int n_in,
                              void* d_out, int out_size, void* d_ws, size_t ws_size,
                              hipStream_t stream) {
  (void)in_sizes; (void)n_in; (void)out_size; (void)ws_size;
  // d_in order (setup_inputs dict insertion order): input_frames, then params
  // en1..en3, de1..de3, hc1, hc2 each as (w, b, g, bb), then 2 scalars.
  const float* input = (const float*)d_in[0];
  struct LP { const float *w, *b, *g, *bb; } P[8];
  for (int l = 0; l < 8; ++l) {
    P[l].w  = (const float*)d_in[1 + 4 * l + 0];
    P[l].b  = (const float*)d_in[1 + 4 * l + 1];
    P[l].g  = (const float*)d_in[1 + 4 * l + 2];
    P[l].bb = (const float*)d_in[1 + 4 * l + 3];
  }

  // Host copies of the layer constants (for repack/BN/zero-fill)
  struct LayerH { int Cin, Cout, KK, CP, Kpad, Npad, M, HW, tr; size_t wOff; };
  LayerH L[8] = {
    {  1,  32, 81,   2,   192,  32, NB * 1024, 1024, 0, 0},
    { 32,  64, 81,  32,  2592,  64, NB *  256,  256, 0, 0},
    { 64, 128, 81,  64,  5184, 128, NB *   64,   64, 0, 0},
    {128,  64, 81, 128, 10368,  64, NB *  225,  225, 1, 0},
    { 64,  32, 81,  64,  5184,  32, NB *  841,  841, 1, 0},
    { 32,  27, 81,  32,  2592,  32, NB * 3249, 3249, 1, 0},
    { 27,   9,  9,  32,   288,  16, NB * 4096, 4096, 0, 0},
    {  9,   1,  9,  16,   160,  16, NB * 4096, 4096, 0, 0},
  };
  size_t wtot = 0;
  for (int l = 0; l < 8; ++l) { L[l].wOff = wtot; wtot += (size_t)L[l].Npad * L[l].Kpad; }

  // ---- carve workspace ----
  char* base = (char*)d_ws; size_t off = 0;
  auto allocB = [&](size_t bytes) -> void* {
    void* p = base + off; off += (bytes + 255) & ~(size_t)255; return p;
  };
  _Float16* wbuf = (_Float16*)allocB(wtot * sizeof(_Float16));
  float*    Cbuf  = (float*)allocB((size_t)1663488 * 4);       // max M*Npad (de3)
  float*    stats = (float*)allocB(256 * 4);
  float*    frame = (float*)allocB((size_t)NB * 4096 * 4);
  float*    hs    = (float*)allocB((size_t)NB * 4096 * 4);
  float*    d3    = (float*)allocB((size_t)NB * 27 * 3249 * 4);
  float*    hid   = (float*)allocB((size_t)NB * 27 * 4096 * 4);
  float*    h2    = (float*)allocB((size_t)NB * 4096 * 4);
  float*    df    = (float*)allocB((size_t)NB * 27 * 4096 * 4);
  float*    offx  = (float*)allocB((size_t)NB * 9 * 4096 * 4);
  float*    offy  = (float*)allocB((size_t)NB * 9 * 4096 * 4);
  float*    img   = (float*)allocB((size_t)NB * 36864 * 4);
  float*    pred  = (float*)allocB((size_t)NB * 4096 * 4);
  _Float16* x0h   = (_Float16*)allocB((size_t)NB * 4096 * 2 * 2);    // NHWC CP=2
  _Float16* a1h   = (_Float16*)allocB((size_t)NB * 1024 * 32 * 2);   // NHWC-32
  _Float16* a2h   = (_Float16*)allocB((size_t)NB * 256 * 64 * 2);    // NHWC-64
  _Float16* a3h   = (_Float16*)allocB((size_t)NB * 64 * 128 * 2);    // NHWC-128
  _Float16* d1h   = (_Float16*)allocB((size_t)NB * 225 * 64 * 2);    // NHWC-64
  _Float16* d2h   = (_Float16*)allocB((size_t)NB * 841 * 32 * 2);    // NHWC-32
  _Float16* hidh  = (_Float16*)allocB((size_t)NB * 4096 * 32 * 2);   // NHWC-32 (27 used)
  _Float16* h1h   = (_Float16*)allocB((size_t)NB * 4096 * 16 * 2);   // NHWC-16 (9 used)

  // ---- weight repack (f32 -> f16 [Npad][Kpad], K = tap*CP + ci) ----
  fill_zero_f16<<<(int)((wtot + 255) / 256), 256, 0, stream>>>(wbuf, (int)wtot);
  for (int l = 0; l < 8; ++l) {
    int total = L[l].Cin * L[l].Cout * L[l].KK;
    int g = (total + 255) / 256;
    if (!L[l].tr)
      repack_fwd_kernel<<<g, 256, 0, stream>>>(P[l].w, wbuf + L[l].wOff,
                                               L[l].Cout, L[l].Cin, L[l].KK, L[l].CP, L[l].Kpad);
    else
      repack_tr_kernel<<<g, 256, 0, stream>>>(P[l].w, wbuf + L[l].wOff,
                                              L[l].Cin, L[l].Cout, L[l].KK, L[l].CP, L[l].Kpad);
  }
  fill_zero_f32<<<(NB * 4096 + 255) / 256, 256, 0, stream>>>(hs, NB * 4096);
  // zero padded-channel NHWC buffers once (pads are never written afterwards)
  fill_zero_f16<<<(NB * 4096 * 32 + 255) / 256, 256, 0, stream>>>(hidh, NB * 4096 * 32);
  fill_zero_f16<<<(NB * 4096 * 16 + 255) / 256, 256, 0, stream>>>(h1h, NB * 4096 * 16);

  auto runBN = [&](int l, float* Yf, _Float16* Yh, int CPH) {
    bn_stats_kernel<<<L[l].Cout, 256, 0, stream>>>(Cbuf, P[l].b, stats, L[l].M, L[l].Npad);
    int tot = L[l].M * L[l].Cout;
    bn_apply_kernel<<<(tot + 255) / 256, 256, 0, stream>>>(
        Cbuf, P[l].b, stats, P[l].g, P[l].bb, Yf, Yh,
        L[l].M, L[l].Npad, L[l].Cout, L[l].HW, CPH);
  };
  auto zeroC = [&](int l) {
    int n = L[l].M * L[l].Npad;
    fill_zero_f32<<<(n + 255) / 256, 256, 0, stream>>>(Cbuf, n);
  };

  const int n64k = NB * 4096;
  const int g64k = (n64k + 255) / 256;
  for (int t = 0; t < NUM_IN + NUM_OUT - 1; ++t) {
    frame_prep_kernel<<<g64k, 256, 0, stream>>>(input, pred, frame, t);
    add2_h_kernel<<<g64k, 256, 0, stream>>>(frame, hs, x0h, n64k);

    // <CP,IH,IW,OH,OW,KDIM,STRIDE,PAD,TR,KPAD,NPAD,NT,SK,LDSW>
    wmma_conv_kernel<  2, 64, 64, 32, 32, 9, 2, 4, 0,   192,  32, 2, 1, 1>
        <<<dim3(1, 1024), 32, 0, stream>>>(x0h, wbuf + L[0].wOff, Cbuf);
    runBN(0, nullptr, a1h, 32);
    zeroC(1);
    wmma_conv_kernel< 32, 32, 32, 16, 16, 9, 2, 4, 0,  2592,  64, 4, 3, 0>
        <<<dim3(3, 256), 32, 0, stream>>>(a1h, wbuf + L[1].wOff, Cbuf);
    runBN(1, nullptr, a2h, 64);
    zeroC(2);
    wmma_conv_kernel< 64, 16, 16,  8,  8, 9, 2, 4, 0,  5184, 128, 8, 6, 0>
        <<<dim3(6, 64), 32, 0, stream>>>(a2h, wbuf + L[2].wOff, Cbuf);
    runBN(2, nullptr, a3h, 128);
    zeroC(3);
    wmma_conv_kernel<128,  8,  8, 15, 15, 9, 2, 4, 1, 10368,  64, 4, 6, 0>
        <<<dim3(6, 225), 32, 0, stream>>>(a3h, wbuf + L[3].wOff, Cbuf);
    runBN(3, nullptr, d1h, 64);
    zeroC(4);
    wmma_conv_kernel< 64, 15, 15, 29, 29, 9, 2, 4, 1,  5184,  32, 2, 6, 1>
        <<<dim3(6, 841), 32, 0, stream>>>(d1h, wbuf + L[4].wOff, Cbuf);
    runBN(4, nullptr, d2h, 32);
    wmma_conv_kernel< 32, 29, 29, 57, 57, 9, 2, 4, 1,  2592,  32, 2, 1, 0>
        <<<dim3(1, 3249), 32, 0, stream>>>(d2h, wbuf + L[5].wOff, Cbuf);
    runBN(5, d3, nullptr, 0);

    upsample_kernel<<<(NB * 27 * 4096 + 255) / 256, 256, 0, stream>>>(d3, hid, hidh);

    wmma_conv_kernel< 32, 64, 64, 64, 64, 3, 1, 1, 0,   288,  16, 1, 1, 1>
        <<<dim3(1, 4096), 32, 0, stream>>>(hidh, wbuf + L[6].wOff, Cbuf);
    runBN(6, nullptr, h1h, 16);
    wmma_conv_kernel< 16, 64, 64, 64, 64, 3, 1, 1, 0,   160,  16, 1, 1, 1>
        <<<dim3(1, 4096), 32, 0, stream>>>(h1h, wbuf + L[7].wOff, Cbuf);
    runBN(7, h2, nullptr, 0);

    add_inplace_kernel<<<g64k, 256, 0, stream>>>(hs, h2, n64k);
    softmax_kernel<<<g64k, 256, 0, stream>>>(hid, df);
    offset_kernel<<<(NB * 9 * 4096 + 255) / 256, 256, 0, stream>>>(frame, df, offx, offy);
    deform_kernel<<<(NB * 9 * 4096 + 255) / 256, 256, 0, stream>>>(frame, offx, offy, img);
    predict_kernel<<<g64k, 256, 0, stream>>>(img, df, pred, (float*)d_out, t);
  }
}
